// ConformanceGNN_58213986730492
// MI455X (gfx1250) — matrix operand order repaired and checked by reading, not compiled, not run on hardware
//
#include <hip/hip_runtime.h>
#include <math.h>

#define H 128
#define NLAYERS 3

typedef __attribute__((ext_vector_type(16))) __bf16 v16bf;
typedef __attribute__((ext_vector_type(8)))  float  v8f;

// ---------------------------------------------------------------------------
// float atomic-max via monotonic integer mapping (works for mixed signs)
// ---------------------------------------------------------------------------
__device__ __forceinline__ void atomic_max_f32(float* addr, float v) {
  int vi = __float_as_int(v);
  if (vi >= 0) atomicMax((int*)addr, vi);
  else         atomicMin((unsigned int*)addr, (unsigned int)vi);
}

// ---------------------------------------------------------------------------
// WMMA GEMM:  C[M,128] = act( [A1 | A2][M,K] @ W[K,128] + bias (+ resid) )
// K = 128 (A2 == nullptr) or 256 (concat).  bf16 operands, f32 accumulate.
// Block = 256 threads = 8 waves; each wave computes a 16x128 strip.
// W is staged in LDS pre-swizzled into the B-operand lane layout.
// ---------------------------------------------------------------------------
__global__ void wmma_gemm_bf16(const float* __restrict__ A1,
                               const float* __restrict__ A2,
                               const float* __restrict__ W,
                               const float* __restrict__ bias,
                               float* __restrict__ Cout,
                               const float* __restrict__ resid,
                               int M, int K, int do_relu)
{
  extern __shared__ __bf16 w_lds[];            // K*128 bf16, B-swizzled
  const int tid  = threadIdx.x;
  const int lane = tid & 31;
  const int wave = tid >> 5;

  // Cooperative load of W into LDS, swizzled so that element
  // i = ((kc*8+nt)*32 + lane)*16 + j  holds W[kc*32 + (j&7) + 8*(lane>=16) + 16*(j>=8)][nt*16 + (lane&15)]
  const int total = K * H;
  for (int i = tid; i < total; i += 256) {
    int j  = i & 15;
    int ln = (i >> 4) & 31;
    int nt = (i >> 9) & 7;
    int kc = i >> 12;
    int k  = kc * 32 + (j & 7) + ((ln >> 4) << 3) + ((j >> 3) << 4);
    int n  = nt * 16 + (ln & 15);
    w_lds[i] = (__bf16)W[(size_t)k * H + n];
  }
  __syncthreads();

  const int m0 = blockIdx.x * 128 + wave * 16;
  if (m0 >= M) return;                         // wave-uniform; no barriers follow

  v8f acc[8];
  #pragma unroll
  for (int t = 0; t < 8; ++t) {
    #pragma unroll
    for (int r = 0; r < 8; ++r) acc[t][r] = 0.0f;
  }

  const int mrow   = m0 + (lane & 15);
  const int mclamp = mrow < M ? mrow : M - 1;  // clamp reads; stores predicated
  const int hiK    = (lane >> 4) << 3;         // 0 or 8 (A layout half-select)
  const int nchunk = K >> 5;

  for (int kc = 0; kc < nchunk; ++kc) {
    int kb = kc << 5;
    const float* src = A1;
    int kloc = kb;
    if (kb >= H) { src = A2; kloc = kb - H; }  // concat second half
    const float* base = src + (size_t)mclamp * H + kloc + hiK;
    if (kc + 1 < nchunk) __builtin_prefetch(base + 32, 0, 1);

    // A tile: 16x32 bf16.  lane element j -> K offset {0..7,16..23} (+hiK)
    float4 f0 = *(const float4*)(base + 0);
    float4 f1 = *(const float4*)(base + 4);
    float4 f2 = *(const float4*)(base + 16);
    float4 f3 = *(const float4*)(base + 20);
    v16bf a;
    a[0]=(__bf16)f0.x; a[1]=(__bf16)f0.y; a[2]=(__bf16)f0.z; a[3]=(__bf16)f0.w;
    a[4]=(__bf16)f1.x; a[5]=(__bf16)f1.y; a[6]=(__bf16)f1.z; a[7]=(__bf16)f1.w;
    a[8]=(__bf16)f2.x; a[9]=(__bf16)f2.y; a[10]=(__bf16)f2.z; a[11]=(__bf16)f2.w;
    a[12]=(__bf16)f3.x; a[13]=(__bf16)f3.y; a[14]=(__bf16)f3.z; a[15]=(__bf16)f3.w;

    const __bf16* wl = w_lds + ((size_t)kc * 4096 + (size_t)lane * 16);
    #pragma unroll
    for (int nt = 0; nt < 8; ++nt) {
      v16bf b = *(const v16bf*)(wl + nt * 512);
      acc[nt] = __builtin_amdgcn_wmma_f32_16x16x32_bf16(
          false, a, false, b, (short)0, acc[nt], false, false);
    }
  }

  // Epilogue: C layout -> VGPR r: m = m0 + r + 8*(lane>=16), n = nt*16 + lane&15
  const int nbase = lane & 15;
  const int rhi   = (lane >> 4) << 3;
  #pragma unroll
  for (int nt = 0; nt < 8; ++nt) {
    int n = nt * 16 + nbase;
    float bv = bias[n];
    #pragma unroll
    for (int r = 0; r < 8; ++r) {
      int m = m0 + r + rhi;
      if (m < M) {
        float v = acc[nt][r] + bv;
        if (resid) v += resid[(size_t)m * H + n];
        if (do_relu) v = v > 0.0f ? v : 0.0f;
        Cout[(size_t)m * H + n] = v;
      }
    }
  }
}

// ---------------------------------------------------------------------------
// Embeddings
// ---------------------------------------------------------------------------
__global__ void embed_place_k(const float* __restrict__ pf, const float* __restrict__ Wpe,
                              const float* __restrict__ bpe, float* __restrict__ ph, int P)
{
  size_t tid = (size_t)blockIdx.x * blockDim.x + threadIdx.x;
  int p = (int)(tid >> 7), h = (int)(tid & 127);
  if (p < P) ph[tid] = pf[p] * Wpe[h] + bpe[h];
}

__global__ void embed_trans_k(const float* __restrict__ tf, const float* __restrict__ Wte,
                              const float* __restrict__ bte, float* __restrict__ th, int T)
{
  size_t tid = (size_t)blockIdx.x * blockDim.x + threadIdx.x;
  int t = (int)(tid >> 7), h = (int)(tid & 127);
  if (t < T) {
    float a = bte[h];
    #pragma unroll
    for (int i = 0; i < 8; ++i) a += tf[(size_t)t * 8 + i] * Wte[i * H + h];
    th[tid] = a;
  }
}

// ---------------------------------------------------------------------------
// Per-node attention score: s[node] = dot(X[node,:], w) + b  (one wave/node)
// ---------------------------------------------------------------------------
__global__ void node_score_k(const float* __restrict__ X, const float* __restrict__ w,
                             const float* __restrict__ bscalar, float* __restrict__ s, int N)
{
  int node = blockIdx.x * 8 + (threadIdx.x >> 5);
  int lane = threadIdx.x & 31;
  if (node >= N) return;
  const float* row = X + (size_t)node * H;
  float acc = 0.0f;
  #pragma unroll
  for (int i = 0; i < 4; ++i) { int h = lane + i * 32; acc += row[h] * w[h]; }
  for (int off = 16; off > 0; off >>= 1) acc += __shfl_down(acc, off, 32);
  if (lane == 0) s[node] = acc + bscalar[0];
}

// ---------------------------------------------------------------------------
// Global softmax over E edges (score gathered from per-node scalar)
// ---------------------------------------------------------------------------
__global__ void init_red_k(float* red) {
  int i = threadIdx.x;
  if (i < 8) red[i] = (i & 1) ? 0.0f : __int_as_float(0xFF800000u); // -inf / 0
}

__global__ void edge_max_k(const float* __restrict__ s, const int* __restrict__ src,
                           int nE, float* red)
{
  __shared__ float sm[256];
  float m = __int_as_float(0xFF800000u);
  for (int e = blockIdx.x * blockDim.x + threadIdx.x; e < nE; e += gridDim.x * blockDim.x)
    m = fmaxf(m, s[src[e]]);
  sm[threadIdx.x] = m; __syncthreads();
  for (int o = 128; o > 0; o >>= 1) {
    if (threadIdx.x < o) sm[threadIdx.x] = fmaxf(sm[threadIdx.x], sm[threadIdx.x + o]);
    __syncthreads();
  }
  if (threadIdx.x == 0) atomic_max_f32(red, sm[0]);
}

__global__ void edge_expsum_k(const float* __restrict__ s, const int* __restrict__ src,
                              int nE, const float* __restrict__ redmax,
                              float* __restrict__ wout, float* __restrict__ redsum)
{
  __shared__ float sm[256];
  float mx = redmax[0];
  float acc = 0.0f;
  for (int e = blockIdx.x * blockDim.x + threadIdx.x; e < nE; e += gridDim.x * blockDim.x) {
    float w = expf(s[src[e]] - mx);
    wout[e] = w;
    acc += w;
  }
  sm[threadIdx.x] = acc; __syncthreads();
  for (int o = 128; o > 0; o >>= 1) {
    if (threadIdx.x < o) sm[threadIdx.x] += sm[threadIdx.x + o];
    __syncthreads();
  }
  if (threadIdx.x == 0) atomicAdd(redsum, sm[0]);
}

// msg[dst[e],:] += (w[e]/sum) * X[src[e],:]   — 128 consecutive threads per edge
__global__ void edge_scatter_k(const float* __restrict__ X, const int* __restrict__ src,
                               const int* __restrict__ dst, const float* __restrict__ wgt,
                               const float* __restrict__ redsum, float* __restrict__ msg,
                               int nE)
{
  size_t tid = (size_t)blockIdx.x * blockDim.x + threadIdx.x;
  int e = (int)(tid >> 7), h = (int)(tid & 127);
  if (e >= nE) return;
  float coef = wgt[e] * (1.0f / redsum[0]);
  atomicAdd(&msg[(size_t)dst[e] * H + h], coef * X[(size_t)src[e] * H + h]);
}

// ---------------------------------------------------------------------------
// Pooling + predictor head
// ---------------------------------------------------------------------------
__global__ void zero_k(float* p, long long n) {
  long long i = (long long)blockIdx.x * blockDim.x + threadIdx.x;
  long long st = (long long)gridDim.x * blockDim.x;
  for (; i < n; i += st) p[i] = 0.0f;
}

__global__ void colsum_k(const float* __restrict__ X, int M, float* __restrict__ out)
{
  int h = threadIdx.x & 127;
  int part = blockIdx.x * 2 + (threadIdx.x >> 7);
  int nparts = gridDim.x * 2;
  float acc = 0.0f;
  for (int r = part; r < M; r += nparts) acc += X[(size_t)r * H + h];
  atomicAdd(&out[h], acc);
}

__global__ void pool_combine_k(const float* __restrict__ csp, const float* __restrict__ cst,
                               const float* __restrict__ Wpp, const float* __restrict__ bpp,
                               const float* __restrict__ Wtp, const float* __restrict__ btp,
                               const float* __restrict__ prefix, const float* __restrict__ Wpr,
                               const float* __restrict__ bpr, float* __restrict__ combined,
                               float invP, float invT)
{
  int j = blockIdx.x * blockDim.x + threadIdx.x;
  if (j < 128) {
    float a = bpp[j];
    for (int k = 0; k < 128; ++k) a += csp[k] * invP * Wpp[k * H + j];
    combined[j] = a;
  } else if (j < 256) {
    int jj = j - 128; float a = btp[jj];
    for (int k = 0; k < 128; ++k) a += cst[k] * invT * Wtp[k * H + jj];
    combined[j] = a;
  } else if (j < 384) {
    int jj = j - 256; float a = bpr[jj];
    for (int k = 0; k < 18; ++k) a += prefix[k] * Wpr[k * H + jj];
    combined[j] = a;
  }
}

__global__ void mlp_head_k(const float* __restrict__ combined,
                           const float* __restrict__ W1, const float* __restrict__ b1,
                           const float* __restrict__ W2, const float* __restrict__ b2,
                           float* __restrict__ h2out)
{
  __shared__ float cmb[384];
  __shared__ float h1[256];
  int j = threadIdx.x;
  for (int i = j; i < 384; i += 256) cmb[i] = combined[i];
  __syncthreads();
  float a = b1[j];
  for (int k = 0; k < 384; ++k) a += cmb[k] * W1[k * 256 + j];
  h1[j] = fmaxf(a, 0.0f);
  __syncthreads();
  if (j < 128) {
    float c = b2[j];
    for (int k = 0; k < 256; ++k) c += h1[k] * W2[k * 128 + j];
    h2out[j] = fmaxf(c, 0.0f);
  }
}

__global__ void logits_k(const float* __restrict__ h2, const float* __restrict__ W3,
                         const float* __restrict__ b3, float* __restrict__ out, int T)
{
  int t = blockIdx.x * blockDim.x + threadIdx.x;
  if (t >= T) return;
  float a = b3[t];
  for (int k = 0; k < 128; ++k) a += h2[k] * W3[(size_t)k * T + t];
  out[t] = 1.0f / (1.0f + expf(-a));
}

// ---------------------------------------------------------------------------
extern "C" void kernel_launch(void* const* d_in, const int* in_sizes, int n_in,
                              void* d_out, int out_size, void* d_ws, size_t ws_size,
                              hipStream_t stream)
{
  const float* pf    = (const float*)d_in[0];
  const float* tf    = (const float*)d_in[1];
  const float* pref  = (const float*)d_in[2];
  const int*   preE  = (const int*)d_in[3];
  const int*   postE = (const int*)d_in[4];
  const float* Wpe = (const float*)d_in[5],  *bpe = (const float*)d_in[6];
  const float* Wte = (const float*)d_in[7],  *bte = (const float*)d_in[8];
  const float* Wpr = (const float*)d_in[9],  *bpr = (const float*)d_in[10];
  const float* p2tW = (const float*)d_in[11], *p2tb = (const float*)d_in[12];
  const float* t2pW = (const float*)d_in[13], *t2pb = (const float*)d_in[14];
  const float* puW  = (const float*)d_in[15], *pub  = (const float*)d_in[16];
  const float* tuW  = (const float*)d_in[17], *tub  = (const float*)d_in[18];
  const float* paW  = (const float*)d_in[19], *pab  = (const float*)d_in[20];
  const float* taW  = (const float*)d_in[21], *tab  = (const float*)d_in[22];
  const float* Wpp = (const float*)d_in[23], *bpp = (const float*)d_in[24];
  const float* Wtp = (const float*)d_in[25], *btp = (const float*)d_in[26];
  const float* W1  = (const float*)d_in[27], *b1  = (const float*)d_in[28];
  const float* W2  = (const float*)d_in[29], *b2  = (const float*)d_in[30];
  const float* W3  = (const float*)d_in[31], *b3  = (const float*)d_in[32];

  const int P = in_sizes[0];
  const int T = in_sizes[32];
  const int E = in_sizes[3] / 2;
  const int* p_src = preE;        const int* p_dst = preE + E;
  const int* t_src = postE;       const int* t_dst = postE + E;

  float* ws = (float*)d_ws;
  size_t off = 0;
  auto alloc = [&](size_t n) { float* p = ws + off; off += n; return p; };
  float* place_h   = alloc((size_t)P * H);
  float* trans_h   = alloc((size_t)T * H);
  float* Xp        = alloc((size_t)P * H);
  float* Xt        = alloc((size_t)T * H);
  float* place_msg = alloc((size_t)P * H);
  float* trans_msg = alloc((size_t)T * H);
  float* sp        = alloc(P);
  float* st        = alloc(T);
  float* wpt       = alloc(E);
  float* wtp       = alloc(E);
  float* red       = alloc(8);     // [0]=max_pt [1]=sum_pt [2]=max_tp [3]=sum_tp
  float* csum      = alloc(256);   // [0:128]=place colsum, [128:256]=trans colsum
  float* combined  = alloc(384);
  float* h2        = alloc(128);

  const int gP = (int)(((size_t)P * H + 255) / 256);
  const int gT = (int)(((size_t)T * H + 255) / 256);
  embed_place_k<<<gP, 256, 0, stream>>>(pf, Wpe, bpe, place_h, P);
  embed_trans_k<<<gT, 256, 0, stream>>>(tf, Wte, bte, trans_h, T);

  const int gemP = (P + 127) / 128, gemT = (T + 127) / 128;
  const size_t sh128 = (size_t)128 * H * sizeof(__bf16);   // 32 KB
  const size_t sh256 = (size_t)256 * H * sizeof(__bf16);   // 64 KB
  const int gE = (int)(((size_t)E * H + 255) / 256);

  for (int l = 0; l < NLAYERS; ++l) {
    // transformed node features (linear-before-gather)
    wmma_gemm_bf16<<<gemP, 256, sh128, stream>>>(place_h, nullptr, p2tW + (size_t)l * H * H,
                                                 p2tb + l * H, Xp, nullptr, P, 128, 0);
    wmma_gemm_bf16<<<gemT, 256, sh128, stream>>>(trans_h, nullptr, t2pW + (size_t)l * H * H,
                                                 t2pb + l * H, Xt, nullptr, T, 128, 0);
    // per-node attention logits
    node_score_k<<<(P + 7) / 8, 256, 0, stream>>>(Xp, taW + l * H, tab + l, sp, P);
    node_score_k<<<(T + 7) / 8, 256, 0, stream>>>(Xt, paW + l * H, pab + l, st, T);
    // global softmax over all E edges (two atomic reductions)
    init_red_k<<<1, 8, 0, stream>>>(red);
    edge_max_k<<<1024, 256, 0, stream>>>(sp, p_src, E, red + 0);
    edge_max_k<<<1024, 256, 0, stream>>>(st, t_src, E, red + 2);
    edge_expsum_k<<<1024, 256, 0, stream>>>(sp, p_src, E, red + 0, wpt, red + 1);
    edge_expsum_k<<<1024, 256, 0, stream>>>(st, t_src, E, red + 2, wtp, red + 3);
    // weighted scatter-add of messages
    zero_k<<<2048, 256, 0, stream>>>(trans_msg, (long long)T * H);
    zero_k<<<2048, 256, 0, stream>>>(place_msg, (long long)P * H);
    edge_scatter_k<<<gE, 256, 0, stream>>>(Xp, p_src, p_dst, wpt, red + 1, trans_msg, E);
    edge_scatter_k<<<gE, 256, 0, stream>>>(Xt, t_src, t_dst, wtp, red + 3, place_msg, E);
    // residual update: h = relu(h + concat(h, msg) @ Wu + bu)  (in place: each
    // block reads/writes only its own 128-row tile)
    wmma_gemm_bf16<<<gemP, 256, sh256, stream>>>(place_h, place_msg, puW + (size_t)l * 2 * H * H,
                                                 pub + l * H, place_h, place_h, P, 256, 1);
    wmma_gemm_bf16<<<gemT, 256, sh256, stream>>>(trans_h, trans_msg, tuW + (size_t)l * 2 * H * H,
                                                 tub + l * H, trans_h, trans_h, T, 256, 1);
  }

  // pooling: mean(X@W+b) == mean(X)@W + b
  zero_k<<<1, 256, 0, stream>>>(csum, 256);
  colsum_k<<<256, 256, 0, stream>>>(place_h, P, csum);
  colsum_k<<<256, 256, 0, stream>>>(trans_h, T, csum + 128);
  pool_combine_k<<<3, 128, 0, stream>>>(csum, csum + 128, Wpp, bpp, Wtp, btp,
                                        pref, Wpr, bpr, combined,
                                        1.0f / (float)P, 1.0f / (float)T);
  mlp_head_k<<<1, 256, 0, stream>>>(combined, W1, b1, W2, b2, h2);
  logits_k<<<(T + 255) / 256, 256, 0, stream>>>(h2, W3, b3, (float*)d_out, T);
}
